// LMAB_29875792511067
// MI455X (gfx1250) — matrix operand order repaired and checked by reading, not compiled, not run on hardware
//
#include <hip/hip_runtime.h>
#include <hip/hip_bf16.h>

// ---------------------------------------------------------------------------
// LMAB block for MI455X (gfx1250, wave32, WMMA).
// Internal layout: NHWC fp32 master tensor in workspace; bf16 activations for
// all 1x1-conv GEMMs via v_wmma_f32_16x16x32_bf16 (f32 accumulate).
// Weight tiles staged into LDS with global_load_async_to_lds_b128 (ASYNCcnt),
// WMMA inner loop software-pipelined against ds_load/global_load latency.
// ---------------------------------------------------------------------------

typedef __attribute__((ext_vector_type(16))) __bf16 v16bf;
typedef __attribute__((ext_vector_type(8)))  float  v8f;

#define NPOS   32768          // 8 * 64 * 64 positions
#define CDIM   128
#define HW     64

__device__ __forceinline__ float gelu_exact(float x) {
    return 0.5f * x * (1.0f + erff(x * 0.70710678118654752f));
}

// ---------------------------------------------------------------------------
// elementwise f32 -> bf16
// ---------------------------------------------------------------------------
__global__ __launch_bounds__(256)
void k_f32_to_bf16(const float* __restrict__ s, __hip_bfloat16* __restrict__ d, int n) {
    int i = blockIdx.x * 256 + threadIdx.x;
    if (i < n) d[i] = __float2bfloat16(s[i]);
}

// ---------------------------------------------------------------------------
// layout converts
// ---------------------------------------------------------------------------
__global__ __launch_bounds__(256)
void k_nchw_to_nhwc(const float* __restrict__ in, float* __restrict__ out) {
    int gid = blockIdx.x * 256 + threadIdx.x;       // NHWC linear, c fastest
    int c   = gid & (CDIM - 1);
    int pos = gid >> 7;
    int b   = pos >> 12;
    int yx  = pos & 4095;
    out[gid] = in[((size_t)(b * CDIM + c) << 12) + yx];
}

__global__ __launch_bounds__(256)
void k_nhwc_to_nchw(const float* __restrict__ in, float* __restrict__ out) {
    int gid = blockIdx.x * 256 + threadIdx.x;       // NCHW linear, x fastest
    int yx  = gid & 4095;
    int c   = (gid >> 12) & (CDIM - 1);
    int b   = gid >> 19;
    out[gid] = in[(size_t)((b << 12) + yx) * CDIM + c];
}

// ---------------------------------------------------------------------------
// LayerNorm over C (NHWC): one wave32 per position, float4 per lane.
// ---------------------------------------------------------------------------
template<bool BF16OUT>
__global__ __launch_bounds__(256)
void k_ln(const float* __restrict__ xf, const float* __restrict__ w,
          const float* __restrict__ b, void* __restrict__ outp) {
    const int lane = threadIdx.x & 31;
    const int wv   = threadIdx.x >> 5;
    const int pos  = blockIdx.x * 8 + wv;

    const float4 v = ((const float4*)(xf + (size_t)pos * CDIM))[lane];
    float s = v.x + v.y + v.z + v.w;
    #pragma unroll
    for (int o = 16; o; o >>= 1) s += __shfl_xor(s, o, 32);
    const float mu = s * (1.0f / 128.0f);

    const float dx = v.x - mu, dy = v.y - mu, dz = v.z - mu, dw = v.w - mu;
    float q = dx * dx + dy * dy + dz * dz + dw * dw;
    #pragma unroll
    for (int o = 16; o; o >>= 1) q += __shfl_xor(q, o, 32);
    const float rs = rsqrtf(q * (1.0f / 128.0f) + 1e-6f);

    const float4 wv4 = ((const float4*)w)[lane];
    const float4 bv4 = ((const float4*)b)[lane];
    const float r0 = dx * rs * wv4.x + bv4.x;
    const float r1 = dy * rs * wv4.y + bv4.y;
    const float r2 = dz * rs * wv4.z + bv4.z;
    const float r3 = dw * rs * wv4.w + bv4.w;

    const int c = lane * 4;
    if (BF16OUT) {
        __hip_bfloat16* o = (__hip_bfloat16*)outp + (size_t)pos * CDIM + c;
        o[0] = __float2bfloat16(r0); o[1] = __float2bfloat16(r1);
        o[2] = __float2bfloat16(r2); o[3] = __float2bfloat16(r3);
    } else {
        float4* o = (float4*)((float*)outp + (size_t)pos * CDIM + c);
        *o = make_float4(r0, r1, r2, r3);
    }
}

// ---------------------------------------------------------------------------
// GEMM: D[M,N] = A[M,128](bf16) x W[N,128]^T(bf16) + bias, K fixed at 128.
// Block = 128 threads = 4 waves; tile 64(M) x 64(N).
// W tile staged via async global->LDS copies (ASYNCcnt); inner loop is a
// 16-step software pipeline (4 K-steps x 4 N-subtiles) with next-fragment
// ds_load/global_load issued while the current WMMA executes.
// ---------------------------------------------------------------------------
template<bool ACCUM>
__global__ __launch_bounds__(128)
void k_gemm_k128(const __hip_bfloat16* __restrict__ A,
                 const __hip_bfloat16* __restrict__ Wt,   // [N,128] row-major
                 const float* __restrict__ bias,
                 float* __restrict__ out, int ldOut) {
    __shared__ __hip_bfloat16 wtile[64 * 128];            // 16 KB

    const int tid  = threadIdx.x;
    const int lane = tid & 31;
    const int wave = tid >> 5;
    const int m0   = blockIdx.x * 64;
    const int n0   = blockIdx.y * 64;

    { // async weight-tile stage: 64 rows x 128 bf16 = 16 KB, 16 B per thread/step
        const char*    gsrc    = (const char*)(Wt + (size_t)n0 * 128);
        const unsigned ldsbase = (unsigned)(uintptr_t)(&wtile[0]); // low 32b = LDS offset
        #pragma unroll
        for (int i = 0; i < 8; ++i) {
            const unsigned off   = (unsigned)(tid + i * 128) * 16u;
            const unsigned laddr = ldsbase + off;
            const char*    gaddr = gsrc + off;
            asm volatile("global_load_async_to_lds_b128 %0, %1, off"
                         :: "v"(laddr), "v"(gaddr) : "memory");
        }
        asm volatile("s_wait_asynccnt 0x0" ::: "memory");
    }
    __syncthreads();

    const int    g    = lane >> 4;                        // half-wave id
    const int    nl   = lane & 15;
    const int    mrow = m0 + wave * 16 + nl;
    const uint4* arow = (const uint4*)(A + (size_t)mrow * 128); // 16 uint4/row

    // B fragment pointer for (kk, t): bf16 k = 8g..8g+7 ([0]) and 16+8g.. ([2]) of row nl+t*16
    auto bptr = [&](int kk, int t) {
        return (const uint4*)(wtile + (t * 16 + nl) * 128 + kk * 32 + g * 8);
    };

    union frag { uint4 u[2]; v16bf v; };
    v8f  acc[4] = {};
    frag af, an, bc, bn;

    af.u[0] = arow[g];                                    // A fragment, kk = 0
    af.u[1] = arow[g + 2];
    an = af;
    bc.u[0] = bptr(0, 0)[0];                              // B fragment, (0,0)
    bc.u[1] = bptr(0, 0)[2];

    #pragma unroll
    for (int i = 0; i < 16; ++i) {                        // i = kk*4 + t
        const int kk = i >> 2;
        const int t  = i & 3;
        if (t == 0 && kk < 3) {                           // A for next K-step, 4 WMMAs early
            an.u[0] = arow[(kk + 1) * 4 + g];
            an.u[1] = arow[(kk + 1) * 4 + g + 2];
            __builtin_prefetch((const void*)(arow + (kk + 1) * 4 + g), 0, 0);
        }
        if (i < 15) {                                     // B for next WMMA, 1 early
            const int j = i + 1;
            bn.u[0] = bptr(j >> 2, j & 3)[0];
            bn.u[1] = bptr(j >> 2, j & 3)[2];
        }
        acc[t] = __builtin_amdgcn_wmma_f32_16x16x32_bf16(
                     false, af.v, false, bc.v, (short)0, acc[t], false, false);
        bc = bn;
        if (t == 3) af = an;
    }

    #pragma unroll
    for (int t = 0; t < 4; ++t) {
        const int   n  = n0 + t * 16 + nl;
        const float bv = bias[n];
        #pragma unroll
        for (int r = 0; r < 8; ++r) {
            const int m   = m0 + wave * 16 + g * 8 + r;   // f32 C/D layout
            const float v = acc[t][r] + bv;
            if (ACCUM) out[(size_t)m * ldOut + n] += v;   // residual into master
            else       out[(size_t)m * ldOut + n]  = v;
        }
    }
}

// ---------------------------------------------------------------------------
// Local neighborhood attention, one thread per (pos, head), online softmax.
// qkv: fp32 [pos][384] (q:0-127, k:128-255, v:256-383). OOB neighbors have
// logit 0 / value 0 and DO participate in softmax (F.unfold zero padding).
// ---------------------------------------------------------------------------
__global__ __launch_bounds__(256)
void k_attn(const float* __restrict__ qkv, __hip_bfloat16* __restrict__ outbf) {
    const int gid  = blockIdx.x * 256 + threadIdx.x;      // NPOS * 8 threads
    const int head = gid & 7;
    const int pos  = gid >> 3;
    const int b    = pos >> 12;
    const int y    = (pos >> 6) & 63;
    const int x    = pos & 63;

    int ks, dil;
    if      (head < 5) { ks = 3; dil = 1; }
    else if (head < 7) { ks = 5; dil = 2; }
    else               { ks = 7; dil = 2; }

    float q[16];
    {
        const float4* qb = (const float4*)(qkv + (size_t)pos * 384 + head * 16);
        #pragma unroll
        for (int d4 = 0; d4 < 4; ++d4) {
            float4 t = qb[d4];
            q[d4 * 4 + 0] = t.x * 0.25f;   // SCALE = 16^-0.5
            q[d4 * 4 + 1] = t.y * 0.25f;
            q[d4 * 4 + 2] = t.z * 0.25f;
            q[d4 * 4 + 3] = t.w * 0.25f;
        }
    }

    float m = -__builtin_inff(), ssum = 0.0f, acc[16];
    #pragma unroll
    for (int d = 0; d < 16; ++d) acc[d] = 0.0f;

    const int hk = ks >> 1;
    for (int i = 0; i < ks; ++i) {
        const int yy = y + (i - hk) * dil;
        for (int j = 0; j < ks; ++j) {
            const int xx = x + (j - hk) * dil;
            float l = 0.0f, vv[16];
            if ((yy >= 0) & (yy < HW) & (xx >= 0) & (xx < HW)) {
                const float4* kb = (const float4*)(qkv +
                    (size_t)((b << 12) + (yy << 6) + xx) * 384 + 128 + head * 16);
                const float4* vb = kb + 32;               // +128 floats
                #pragma unroll
                for (int d4 = 0; d4 < 4; ++d4) {
                    float4 kv = kb[d4];
                    float4 v4 = vb[d4];
                    l += q[d4 * 4 + 0] * kv.x + q[d4 * 4 + 1] * kv.y
                       + q[d4 * 4 + 2] * kv.z + q[d4 * 4 + 3] * kv.w;
                    vv[d4 * 4 + 0] = v4.x; vv[d4 * 4 + 1] = v4.y;
                    vv[d4 * 4 + 2] = v4.z; vv[d4 * 4 + 3] = v4.w;
                }
            } else {
                #pragma unroll
                for (int d = 0; d < 16; ++d) vv[d] = 0.0f;
            }
            const float mn   = fmaxf(m, l);
            const float corr = __expf(m - mn);
            const float p    = __expf(l - mn);
            ssum = ssum * corr + p;
            #pragma unroll
            for (int d = 0; d < 16; ++d) acc[d] = acc[d] * corr + p * vv[d];
            m = mn;
        }
    }

    const float inv = 1.0f / ssum;
    __hip_bfloat16* o = outbf + (size_t)pos * CDIM + head * 16;
    #pragma unroll
    for (int d = 0; d < 16; ++d) o[d] = __float2bfloat16(acc[d] * inv);
}

// ---------------------------------------------------------------------------
// MSDWConv + star gate for MSConvStar: h fp32 [pos][256] -> bf16 [pos][128].
// channel groups of 64 use k = 1,3,5,7; y = h + dw(h); out = gelu(x1)*x2.
// ---------------------------------------------------------------------------
__device__ __forceinline__ float dw_accum(const float* __restrict__ h, int b, int y,
                                          int x, int ch, const float* __restrict__ w,
                                          int k, int cc) {
    const int hk = k >> 1;
    float r = 0.0f;
    for (int i = 0; i < k; ++i) {
        const int yy = y + i - hk;
        if (yy < 0 || yy >= HW) continue;
        for (int j = 0; j < k; ++j) {
            const int xx = x + j - hk;
            if (xx < 0 || xx >= HW) continue;
            r += h[(size_t)((b << 12) + (yy << 6) + xx) * 256 + ch]
               * w[(cc * k + i) * k + j];
        }
    }
    return r;
}

__global__ __launch_bounds__(256)
void k_dwgate(const float* __restrict__ h,
              const float* __restrict__ w0, const float* __restrict__ b0,
              const float* __restrict__ w1, const float* __restrict__ b1,
              const float* __restrict__ w2, const float* __restrict__ b2,
              const float* __restrict__ w3, const float* __restrict__ b3,
              __hip_bfloat16* __restrict__ g) {
    const int gid = blockIdx.x * 256 + threadIdx.x;       // NPOS * 128 threads
    const int c   = gid & 127;
    const int pos = gid >> 7;
    const int b   = pos >> 12;
    const int y   = (pos >> 6) & 63;
    const int x   = pos & 63;
    const int cc  = c & 63;

    const float hc = h[(size_t)pos * 256 + c];
    const float hd = h[(size_t)pos * 256 + 128 + c];

    float y1, y2;
    if (c < 64) y1 = b0[cc] + dw_accum(h, b, y, x, c,        w0, 1, cc);
    else        y1 = b1[cc] + dw_accum(h, b, y, x, c,        w1, 3, cc);
    if (c < 64) y2 = b2[cc] + dw_accum(h, b, y, x, 128 + c,  w2, 5, cc);
    else        y2 = b3[cc] + dw_accum(h, b, y, x, 128 + c,  w3, 7, cc);

    const float x1 = hc + y1;
    const float x2 = hd + y2;
    g[(size_t)pos * CDIM + c] = __float2bfloat16(gelu_exact(x1) * x2);
}

// ---------------------------------------------------------------------------
// SMA depthwise 3x3 (dilated) + GELU, fp32 NHWC [pos][128] -> same.
// ---------------------------------------------------------------------------
__global__ __launch_bounds__(256)
void k_sma_dw(const float* __restrict__ in, const float* __restrict__ w,
              const float* __restrict__ bs, float* __restrict__ out, int dil) {
    const int gid = blockIdx.x * 256 + threadIdx.x;
    const int c   = gid & 127;
    const int pos = gid >> 7;
    const int b   = pos >> 12;
    const int y   = (pos >> 6) & 63;
    const int x   = pos & 63;

    float r = bs[c];
    #pragma unroll
    for (int i = 0; i < 3; ++i) {
        const int yy = y + (i - 1) * dil;
        if (yy < 0 || yy >= HW) continue;
        #pragma unroll
        for (int j = 0; j < 3; ++j) {
            const int xx = x + (j - 1) * dil;
            if (xx < 0 || xx >= HW) continue;
            r += in[(size_t)((b << 12) + (yy << 6) + xx) * CDIM + c]
               * w[(c * 3 + i) * 3 + j];
        }
    }
    out[(size_t)pos * CDIM + c] = gelu_exact(r);
}

// ---------------------------------------------------------------------------
// host-side launcher
// ---------------------------------------------------------------------------
extern "C" void kernel_launch(void* const* d_in, const int* in_sizes, int n_in,
                              void* d_out, int out_size, void* d_ws, size_t ws_size,
                              hipStream_t stream) {
    (void)in_sizes; (void)n_in; (void)out_size; (void)ws_size;

    const float* x       = (const float*)d_in[0];
    const float* ln_w[4] = {(const float*)d_in[1], (const float*)d_in[3],
                            (const float*)d_in[5], (const float*)d_in[7]};
    const float* ln_b[4] = {(const float*)d_in[2], (const float*)d_in[4],
                            (const float*)d_in[6], (const float*)d_in[8]};
    const float* qkv_w   = (const float*)d_in[9];
    const float* qkv_b   = (const float*)d_in[10];
    const float* proj_w  = (const float*)d_in[11];
    const float* proj_b  = (const float*)d_in[12];
    const float* fc1_w[2] = {(const float*)d_in[13], (const float*)d_in[25]};
    const float* fc1_b[2] = {(const float*)d_in[14], (const float*)d_in[26]};
    const float* dww[2][4] = {
        {(const float*)d_in[15], (const float*)d_in[16], (const float*)d_in[17], (const float*)d_in[18]},
        {(const float*)d_in[27], (const float*)d_in[28], (const float*)d_in[29], (const float*)d_in[30]}};
    const float* dwb[2][4] = {
        {(const float*)d_in[19], (const float*)d_in[20], (const float*)d_in[21], (const float*)d_in[22]},
        {(const float*)d_in[31], (const float*)d_in[32], (const float*)d_in[33], (const float*)d_in[34]}};
    const float* fc2_w[2] = {(const float*)d_in[23], (const float*)d_in[35]};
    const float* fc2_b[2] = {(const float*)d_in[24], (const float*)d_in[36]};
    const float* sma_w[3] = {(const float*)d_in[37], (const float*)d_in[38], (const float*)d_in[39]};
    const float* sma_b[3] = {(const float*)d_in[40], (const float*)d_in[41], (const float*)d_in[42]};
    const float* pw_w = (const float*)d_in[43];
    const float* pw_b = (const float*)d_in[44];

    char* ws = (char*)d_ws;
    float*          xf = (float*)ws;                              // 16 MB master NHWC
    __hip_bfloat16* ab = (__hip_bfloat16*)(ws + (16ull << 20));   //  8 MB bf16 A
    float*          qk = (float*)(ws + (24ull << 20));            // 48 MB (qkv / sma ping)
    float*          hb = (float*)(ws + (72ull << 20));            // 32 MB (fc1 out / sma pong)
    __hip_bfloat16* wb = (__hip_bfloat16*)(ws + (104ull << 20));  // bf16 packed weights
    __hip_bfloat16* Wqkv  = wb;
    __hip_bfloat16* Wproj = wb + 49152;
    __hip_bfloat16* Wfc1a = wb + 65536;
    __hip_bfloat16* Wfc2a = wb + 98304;
    __hip_bfloat16* Wfc1b = wb + 114688;
    __hip_bfloat16* Wfc2b = wb + 147456;
    __hip_bfloat16* Wpw   = wb + 163840;

    auto cvt = [&](const float* s, __hip_bfloat16* d, int n) {
        k_f32_to_bf16<<<(n + 255) / 256, 256, 0, stream>>>(s, d, n);
    };
    cvt(qkv_w,    Wqkv,  384 * 128);
    cvt(proj_w,   Wproj, 128 * 128);
    cvt(fc1_w[0], Wfc1a, 256 * 128);
    cvt(fc2_w[0], Wfc2a, 128 * 128);
    cvt(fc1_w[1], Wfc1b, 256 * 128);
    cvt(fc2_w[1], Wfc2b, 128 * 128);
    cvt(pw_w,     Wpw,   128 * 128);

    const int EBLK = NPOS * CDIM / 256;    // 16384 blocks for per-(pos,c) kernels
    k_nchw_to_nhwc<<<EBLK, 256, 0, stream>>>(x, xf);

    const dim3 gqkv(NPOS / 64, 6), g256(NPOS / 64, 4), g128(NPOS / 64, 2);

    // ---- block 1: x += attn_proj(attention(qkv(LN1(x)))) ----
    k_ln<true><<<NPOS / 8, 256, 0, stream>>>(xf, ln_w[0], ln_b[0], ab);
    k_gemm_k128<false><<<gqkv, 128, 0, stream>>>(ab, Wqkv, qkv_b, qk, 384);
    k_attn<<<NPOS * 8 / 256, 256, 0, stream>>>(qk, ab);
    k_gemm_k128<true><<<g128, 128, 0, stream>>>(ab, Wproj, proj_b, xf, 128);

    // ---- block 2: x += MSConvStar(LN2(x), mlp1) ----
    k_ln<true><<<NPOS / 8, 256, 0, stream>>>(xf, ln_w[1], ln_b[1], ab);
    k_gemm_k128<false><<<g256, 128, 0, stream>>>(ab, Wfc1a, fc1_b[0], hb, 256);
    k_dwgate<<<EBLK, 256, 0, stream>>>(hb, dww[0][0], dwb[0][0], dww[0][1], dwb[0][1],
                                       dww[0][2], dwb[0][2], dww[0][3], dwb[0][3], ab);
    k_gemm_k128<true><<<g128, 128, 0, stream>>>(ab, Wfc2a, fc2_b[0], xf, 128);

    // ---- block 3: x += SMA(LN3(x)) ----
    k_ln<false><<<NPOS / 8, 256, 0, stream>>>(xf, ln_w[2], ln_b[2], qk);
    k_sma_dw<<<EBLK, 256, 0, stream>>>(qk, sma_w[0], sma_b[0], hb, 1);
    k_sma_dw<<<EBLK, 256, 0, stream>>>(hb, sma_w[1], sma_b[1], qk, 2);
    k_sma_dw<<<EBLK, 256, 0, stream>>>(qk, sma_w[2], sma_b[2], hb, 3);
    k_f32_to_bf16<<<EBLK, 256, 0, stream>>>(hb, ab, NPOS * CDIM);
    k_gemm_k128<true><<<g128, 128, 0, stream>>>(ab, Wpw, pw_b, xf, 128);

    // ---- block 4: x += MSConvStar(LN4(x), mlp2) ----
    k_ln<true><<<NPOS / 8, 256, 0, stream>>>(xf, ln_w[3], ln_b[3], ab);
    k_gemm_k128<false><<<g256, 128, 0, stream>>>(ab, Wfc1b, fc1_b[1], hb, 256);
    k_dwgate<<<EBLK, 256, 0, stream>>>(hb, dww[1][0], dwb[1][0], dww[1][1], dwb[1][1],
                                       dww[1][2], dwb[1][2], dww[1][3], dwb[1][3], ab);
    k_gemm_k128<true><<<g128, 128, 0, stream>>>(ab, Wfc2b, fc2_b[1], xf, 128);

    k_nhwc_to_nchw<<<EBLK, 256, 0, stream>>>(xf, (float*)d_out);
}
